// Attention_58755152609998
// MI455X (gfx1250) — matrix operand order repaired and checked by reading, not compiled, not run on hardware
//
#include <hip/hip_runtime.h>
#include <hip/hip_bf16.h>

// ---------- CDNA5 WMMA types ----------
typedef __attribute__((ext_vector_type(16))) unsigned short v16u;
typedef __attribute__((ext_vector_type(16))) __bf16         v16bf;
typedef __attribute__((ext_vector_type(2)))  __bf16         v2bf;
typedef __attribute__((ext_vector_type(8)))  float          v8f;

union frag {                       // one WMMA A/B fragment (32 bytes)
  v16u           u;
  v16bf          b;
  unsigned int   w[8];
  uint4          q[2];
};

static __device__ __forceinline__ unsigned short f2bf(float f) {
  __bf16 h = (__bf16)f;            // native RNE convert
  union { __bf16 b; unsigned short u; } c; c.b = h; return c.u;
}

// pack two floats -> two bf16 in one dword (v_cvt_pk_bf16_f32, codegen-confirmed)
static __device__ __forceinline__ unsigned int pk_bf16(float a, float b) {
#if __has_builtin(__builtin_amdgcn_cvt_pk_bf16_f32)
  v2bf v = __builtin_amdgcn_cvt_pk_bf16_f32(a, b);
  union { v2bf v; unsigned int u; } c; c.v = v; return c.u;
#else
  return (unsigned int)f2bf(a) | ((unsigned int)f2bf(b) << 16);
#endif
}

// lane^1 swap: single VALU v_mov_b32 dpp8:[1,0,3,2,5,4,7,6] (codegen-confirmed)
static __device__ __forceinline__ float swap1(float v) {
#if __has_builtin(__builtin_amdgcn_mov_dpp8)
  const unsigned int SEL = (1u << 0) | (0u << 3) | (3u << 6) | (2u << 9) |
                           (5u << 12) | (4u << 15) | (7u << 18) | (6u << 21);
  return __int_as_float(__builtin_amdgcn_mov_dpp8(__float_as_int(v), SEL));
#else
  return __shfl_xor(v, 1);
#endif
}

static __device__ __forceinline__ v16bf as_bf(v16u u) {
  union { v16u u; v16bf b; } c; c.u = u; return c.b;
}

static __device__ __forceinline__ v8f wmma_bf16(v16bf a, v16bf b, v8f c) {
  return __builtin_amdgcn_wmma_f32_16x16x32_bf16(false, a, false, b, (short)0, c, false, false);
}

// ---------- sizes ----------
// B=32, grid=8, psize=8, N=64 patches, P=64 pixels, C=256, h=8, d=32, D=2048
// tokens = 131072 ; per-(b,n) qkv flat = 64*768 = 49152

// ---------- kernel 0: weight transpose + bf16 convert ----------
__global__ __launch_bounds__(256)
void prep_weights(const float* __restrict__ wqkv, const float* __restrict__ wproj,
                  unsigned short* __restrict__ wqkvT, unsigned short* __restrict__ wprojT) {
  int stride = gridDim.x * blockDim.x;
  int idx = blockIdx.x * blockDim.x + threadIdx.x;
  for (int i = idx; i < 768 * 256; i += stride) {        // wqkvT[c][k] = wqkv[k][c]
    int c = i >> 8, k = i & 255;
    wqkvT[i] = f2bf(wqkv[k * 768 + c]);
  }
  for (int i = idx; i < 256 * 256; i += stride) {        // wprojT[c][k] = wproj[k][c]
    int c = i >> 8, k = i & 255;
    wprojT[i] = f2bf(wproj[k * 256 + c]);
  }
}

// ---------- kernel 1: QKV GEMM (gathered x @ w_qkv -> bf16 qkv[token][768]) ----------
// wave tile: 16 tokens x 128 cols (8 accumulators); A converted once per chunk
__global__ __launch_bounds__(256)
void qkv_gemm(const float* __restrict__ x, const unsigned short* __restrict__ wqkvT,
              unsigned short* __restrict__ qkv) {
  const int lane = threadIdx.x & 31;
  const int wv   = threadIdx.x >> 5;
  const int hi   = lane >> 4;
  const int lo   = lane & 15;
  const int tt   = blockIdx.x * 8 + wv;          // token tile (16 tokens)
  const int colbase = blockIdx.y * 128;

  // A-row gather address: token -> spatial x row (patch partition)
  int tok = tt * 16 + lo;
  int b = tok >> 12, n = (tok >> 6) & 63, p = tok & 63;
  int r  = ((n >> 3) << 3) | (p >> 3);
  int cc = ((n & 7) << 3) | (p & 7);
  const float* xrow = x + (((b * 64 + r) * 64 + cc) << 8);

  v8f acc[8] = {};
  for (int kc = 0; kc < 256; kc += 32) {
    // A fragment (16x32 bf16): lane row = lo; K = hi*16+{0..15}
    const float* xa = xrow + kc + hi * 16;
    float4 fa = *(const float4*)(xa);
    float4 fb = *(const float4*)(xa + 4);
    float4 fc = *(const float4*)(xa + 8);
    float4 fd = *(const float4*)(xa + 12);
    frag af;
    af.w[0] = pk_bf16(fa.x, fa.y);  af.w[1] = pk_bf16(fa.z, fa.w);
    af.w[2] = pk_bf16(fb.x, fb.y);  af.w[3] = pk_bf16(fb.z, fb.w);
    af.w[4] = pk_bf16(fc.x, fc.y);  af.w[5] = pk_bf16(fc.z, fc.w);
    af.w[6] = pk_bf16(fd.x, fd.y);  af.w[7] = pk_bf16(fd.z, fd.w);
#pragma unroll
    for (int t = 0; t < 8; ++t) {
      // B fragment (32x16): lane col = colbase+t*16+lo, K = kc + hi*16 + {0..15}
      const v16u* bp = (const v16u*)(wqkvT + ((colbase + t * 16 + lo) << 8) + kc + hi * 16);
      acc[t] = wmma_bf16(af.b, as_bf(*bp), acc[t]);
    }
  }
  // paired stores: even lanes store tile tp, odd lanes tile tp+1, 32-bit each
#pragma unroll
  for (int tp = 0; tp < 8; tp += 2) {
    const int odd = lane & 1;
    const int colEven = colbase + (tp + odd) * 16 + lo - odd;   // even column of pair
#pragma unroll
    for (int v = 0; v < 8; ++v) {
      float m0 = acc[tp][v];
      float m1 = acc[tp + 1][v];
      float p0 = swap1(m0);
      float p1 = swap1(m1);
      unsigned int val = odd ? pk_bf16(p1, m1) : pk_bf16(m0, p0);
      int row = tt * 16 + v + hi * 8;            // C/D layout: M = v + hi*8
      *(unsigned int*)&qkv[row * 768 + colEven] = val;
    }
  }
}

// ---------- kernel 2: attention per (b,h) ----------
__global__ __launch_bounds__(256)
void attn_kernel(const unsigned short* __restrict__ qkv, unsigned short* __restrict__ obuf) {
  __shared__ float          S[64][64];           // 16 KB logits
  __shared__ unsigned short P[64][64];           //  8 KB probs (bf16)
  __shared__ unsigned short Qc[2][64][32];       //  8 KB double-buffered Q chunk
  __shared__ unsigned short Kc[2][64][32];       //  8 KB double-buffered K chunk
  __shared__ unsigned short VcT[2][2][16][64];   //  8 KB double-buffered V, transposed [col][m]

  const int bh = blockIdx.x;
  const int b = bh >> 3, h = bh & 7;
  const int lane = threadIdx.x & 31;
  const int wv   = threadIdx.x >> 5;
  const int hi = lane >> 4, lo = lane & 15;

  // per-(b,n) base into qkv; Q at +0, K at +16384, V at +32768
  const int base_bn = (b * 64) * 49152 + h * 2048;

  // ---- Phase A: S = Q K^T (double-buffered chunk staging, 1 barrier/iter) ----
  const int sr   = threadIdx.x >> 2;             // staging row 0..63
  const int sseg = threadIdx.x & 3;
  const unsigned short* qk_src = qkv + base_bn + sr * 49152 + sseg * 8;

  *(uint4*)&Qc[0][sr][sseg * 8] = *(const uint4*)(qk_src);
  *(uint4*)&Kc[0][sr][sseg * 8] = *(const uint4*)(qk_src + 16384);
  __syncthreads();

  const int rb = wv >> 1;                        // row block 0..3
  const int cpair = wv & 1;                      // cols {0,16} or {32,48}
  v8f acc0 = {}, acc1 = {};

  for (int kc = 0; kc < 64; ++kc) {
    int cur = kc & 1;
    if (kc + 1 < 64) {
      const unsigned short* s2 = qk_src + (kc + 1) * 32;
      *(uint4*)&Qc[cur ^ 1][sr][sseg * 8] = *(const uint4*)(s2);
      *(uint4*)&Kc[cur ^ 1][sr][sseg * 8] = *(const uint4*)(s2 + 16384);
    }
    frag af, bf;
    int ar = rb * 16 + lo;
    af.q[0] = *(const uint4*)&Qc[cur][ar][hi * 16];
    af.q[1] = *(const uint4*)&Qc[cur][ar][hi * 16 + 8];
    int c0 = (cpair * 2 + 0) * 16 + lo;          // B = K^T : B[k][col] = Kc[col][k]
    bf.q[0] = *(const uint4*)&Kc[cur][c0][hi * 16];
    bf.q[1] = *(const uint4*)&Kc[cur][c0][hi * 16 + 8];
    acc0 = wmma_bf16(af.b, bf.b, acc0);
    int c1 = (cpair * 2 + 1) * 16 + lo;
    bf.q[0] = *(const uint4*)&Kc[cur][c1][hi * 16];
    bf.q[1] = *(const uint4*)&Kc[cur][c1][hi * 16 + 8];
    acc1 = wmma_bf16(af.b, bf.b, acc1);
    __syncthreads();
  }
#pragma unroll
  for (int v = 0; v < 8; ++v) {
    S[rb * 16 + v + hi * 8][(cpair * 2 + 0) * 16 + lo] = acc0[v];
    S[rb * 16 + v + hi * 8][(cpair * 2 + 1) * 16 + lo] = acc1[v];
  }
  __syncthreads();

  // ---- Phase B: softmax rows (threads 0..63, one row each, float4-vectorized) ----
  if (threadIdx.x < 64) {
    const float scale = 0.022097086912079612f;   // (d*P)^-0.5 = 2048^-0.5
    int rr = threadIdx.x;
    float4* srow = (float4*)&S[rr][0];
    float mx = -3.4e38f;
#pragma unroll
    for (int c = 0; c < 16; ++c) {
      float4 v = srow[c];
      mx = fmaxf(mx, fmaxf(fmaxf(v.x, v.y), fmaxf(v.z, v.w)));
    }
    float sum = 0.f;
#pragma unroll
    for (int c = 0; c < 16; ++c) {
      float4 v = srow[c];
      v.x = __expf((v.x - mx) * scale);
      v.y = __expf((v.y - mx) * scale);
      v.z = __expf((v.z - mx) * scale);
      v.w = __expf((v.w - mx) * scale);
      srow[c] = v;
      sum += (v.x + v.y) + (v.z + v.w);
    }
    float inv = 1.0f / sum;
    unsigned int* prow = (unsigned int*)&P[rr][0];
#pragma unroll
    for (int c = 0; c < 16; ++c) {
      float4 v = srow[c];
      prow[c * 2 + 0] = pk_bf16(v.x * inv, v.y * inv);
      prow[c * 2 + 1] = pk_bf16(v.z * inv, v.w * inv);
    }
  }
  __syncthreads();

  // ---- Phase C: O = P V (8 waves = 4 row blocks x 2 col streams) ----
  const int rb2 = wv & 3, half = wv >> 2;
  v16bf pa[2];
  {
    int ar = rb2 * 16 + lo;
#pragma unroll
    for (int ch = 0; ch < 2; ++ch) {
      frag af;
      af.q[0] = *(const uint4*)&P[ar][ch * 32 + hi * 16];
      af.q[1] = *(const uint4*)&P[ar][ch * 32 + hi * 16 + 8];
      pa[ch] = af.b;
    }
  }
  const int vbuf = threadIdx.x >> 7;             // which col-tile of the pair
  const int vm   = (threadIdx.x >> 1) & 63;      // V row m
  const int vseg = threadIdx.x & 1;
  const unsigned short* vsrc = qkv + base_bn + 32768 + vm * 49152 + vseg * 8;

  auto stageV = [&](int pp, int bb) {
    uint4 d = *(const uint4*)(vsrc + pp * 32 + vbuf * 16);
    unsigned short* dst = &VcT[bb][vbuf][0][vm];
    int c0 = vseg * 8;
    dst[(c0 + 0) * 64] = (unsigned short)(d.x);
    dst[(c0 + 1) * 64] = (unsigned short)(d.x >> 16);
    dst[(c0 + 2) * 64] = (unsigned short)(d.y);
    dst[(c0 + 3) * 64] = (unsigned short)(d.y >> 16);
    dst[(c0 + 4) * 64] = (unsigned short)(d.z);
    dst[(c0 + 5) * 64] = (unsigned short)(d.z >> 16);
    dst[(c0 + 6) * 64] = (unsigned short)(d.w);
    dst[(c0 + 7) * 64] = (unsigned short)(d.w >> 16);
  };

  stageV(0, 0);
  __syncthreads();
  for (int it = 0; it < 64; ++it) {              // 128 col tiles, 2 per iter
    int cur = it & 1;
    if (it + 1 < 64) stageV(it + 1, cur ^ 1);
    v8f acc = {};
#pragma unroll
    for (int ch = 0; ch < 2; ++ch) {             // K-dim m: two chunks of 32
      frag bf;
      bf.q[0] = *(const uint4*)&VcT[cur][half][lo][ch * 32 + hi * 16];
      bf.q[1] = *(const uint4*)&VcT[cur][half][lo][ch * 32 + hi * 16 + 8];
      acc = wmma_bf16(pa[ch], bf.b, acc);
    }
    int dcolEven = (it * 2 + half) * 16 + (lo & ~1);
#pragma unroll
    for (int v = 0; v < 8; ++v) {
      float m  = acc[v];
      float pr = swap1(m);
      if (!(lane & 1)) {
        int n2 = rb2 * 16 + v + hi * 8;
        // o_fc layout == per-(b,n) offset h*2048 + dcol (the torch reshape chain)
        *(unsigned int*)&obuf[((b * 64 + n2) * 8 + h) * 2048 + dcolEven] = pk_bf16(m, pr);
      }
    }
    __syncthreads();
  }
}

// ---------- kernel 3: output projection + bias + patch reassembly ----------
// wave tile: 16 tokens x 128 cols (8 accumulators); obuf read 2x instead of 4x
__global__ __launch_bounds__(256)
void proj_gemm(const unsigned short* __restrict__ obuf, const unsigned short* __restrict__ wprojT,
               const float* __restrict__ bias, float* __restrict__ out) {
  const int lane = threadIdx.x & 31;
  const int wv   = threadIdx.x >> 5;
  const int hi = lane >> 4, lo = lane & 15;
  const int tt = blockIdx.x * 8 + wv;
  const int colbase = blockIdx.y * 128;

  const unsigned short* arow = obuf + (tt * 16 + lo) * 256;
  v8f acc[8] = {};
  for (int kc = 0; kc < 256; kc += 32) {
    frag af;
    const unsigned short* ap = arow + kc + hi * 16;
    af.q[0] = *(const uint4*)(ap);
    af.q[1] = *(const uint4*)(ap + 8);
#pragma unroll
    for (int t = 0; t < 8; ++t) {
      const v16u* bp = (const v16u*)(wprojT + ((colbase + t * 16 + lo) << 8) + kc + hi * 16);
      acc[t] = wmma_bf16(af.b, as_bf(*bp), acc[t]);
    }
  }
#pragma unroll
  for (int t = 0; t < 8; ++t) {
    int c = colbase + t * 16 + lo;
    float bv = bias[c];
#pragma unroll
    for (int v = 0; v < 8; ++v) {
      int tok = tt * 16 + v + hi * 8;
      int b = tok >> 12, n = (tok >> 6) & 63, p = tok & 63;
      int r  = ((n >> 3) << 3) | (p >> 3);     // H index
      int cc = ((n & 7) << 3) | (p & 7);       // W index
      out[(((b * 64 + r) * 64 + cc) << 8) + c] = acc[t][v] + bv;
    }
  }
}

// ---------- launcher ----------
extern "C" void kernel_launch(void* const* d_in, const int* in_sizes, int n_in,
                              void* d_out, int out_size, void* d_ws, size_t ws_size,
                              hipStream_t stream) {
  (void)in_sizes; (void)n_in; (void)out_size; (void)ws_size;
  const float* x      = (const float*)d_in[0];
  const float* w_qkv  = (const float*)d_in[1];
  const float* w_proj = (const float*)d_in[2];
  const float* b_proj = (const float*)d_in[3];
  float* out = (float*)d_out;

  unsigned short* qkv    = (unsigned short*)d_ws;                               // 192 MiB
  unsigned short* obuf   = (unsigned short*)((char*)d_ws + (size_t)201326592);  //  64 MiB
  unsigned short* wqkvT  = (unsigned short*)((char*)d_ws + (size_t)268435456);  // 384 KiB
  unsigned short* wprojT = wqkvT + 768 * 256;                                   // 128 KiB

  prep_weights<<<64, 256, 0, stream>>>(w_qkv, w_proj, wqkvT, wprojT);
  qkv_gemm<<<dim3(1024, 6), 256, 0, stream>>>(x, wqkvT, qkv);
  attn_kernel<<<256, 256, 0, stream>>>(qkv, obuf);
  proj_gemm<<<dim3(1024, 2), 256, 0, stream>>>(obuf, wprojT, b_proj, out);
}
